// MultiheadAttention_19138374271344
// MI455X (gfx1250) — matrix operand order repaired and checked by reading, not compiled, not run on hardware
//
// MI455X (gfx1250) fused MHA + RoPE, flash-attention style.
// fp32 in/out; internally bf16 WMMA (v_wmma_f32_16x16x32_bf16, fp32 accum).
// Round 3: ping-pong (unroll-by-2) k-loop in the projection GEMMs -- removes
// the v_dual_mov register rotation and the WMMA WAR-hazard v_nops it caused,
// while keeping fragment loads one k-step ahead of the WMMA bursts.
// Workspace needed: ~44 MB (L2-resident on MI455X's 192MB L2).
#include <hip/hip_runtime.h>

typedef __attribute__((ext_vector_type(16))) __bf16   v16bf;
typedef __attribute__((ext_vector_type(8)))  float    v8f;
typedef __attribute__((ext_vector_type(4)))  unsigned u32x4;

#define DEVINL __device__ __forceinline__

static constexpr int T_  = 2048;
static constexpr int B_  = 4;
static constexpr int E_  = 512;
static constexpr int H_  = 8;
static constexpr int HD_ = 64;
static constexpr int N_  = T_ * B_;   // 8192 tokens
static constexpr int BH_ = B_ * H_;   // 32 batch-heads

union FragU { v16bf v; u32x4 q[2]; };

// A operand: 16x32 bf16, row m = lane%16. ISA layout: half=lane/16,
// VGPR0..3 hold K = 8*half + 0..7 (pairs), VGPR4..7 hold K = 16 + 8*half + 0..7.
DEVINL v16bf load_frag_a(const __bf16* base, int stride, int lane) {
  const int m = lane & 15, hf = lane >> 4;
  const __bf16* p = base + m * stride + 8 * hf;
  FragU f;
  f.q[0] = *(const u32x4*)(p);
  f.q[1] = *(const u32x4*)(p + 16);
  return f.v;
}

// B operand: 32x16 bf16, column n = lane%16; same K layout as A:
// elems 0..7 hold K = 8*half + 0..7, elems 8..15 hold K = 16 + 8*half + 0..7. Source row-major [n][k].
DEVINL v16bf load_frag_b(const __bf16* base, int stride, int lane) {
  const int n = lane & 15, hf = lane >> 4;
  const __bf16* p = base + n * stride + 8 * hf;
  FragU f;
  f.q[0] = *(const u32x4*)(p);
  f.q[1] = *(const u32x4*)(p + 16);
  return f.v;
}

DEVINL v8f wmma_bf16(v16bf a, v16bf b, v8f c) {
  return __builtin_amdgcn_wmma_f32_16x16x32_bf16(false, a, false, b, (short)0, c,
                                                 false, false);
}

DEVINL v8f vzero8() {
  v8f v = {0.f, 0.f, 0.f, 0.f, 0.f, 0.f, 0.f, 0.f};
  return v;
}

DEVINL void load_a2(v16bf a[2], const __bf16* Abase, int kk, int lane) {
#pragma unroll
  for (int mt = 0; mt < 2; ++mt)
    a[mt] = load_frag_a(Abase + mt * 16 * E_ + kk, E_, lane);
}

DEVINL void load_b4(v16bf b[4], const __bf16* Bbase, int kk, int lane) {
#pragma unroll
  for (int j = 0; j < 4; ++j)
    b[j] = load_frag_b(Bbase + j * 16 * E_ + kk, E_, lane);
}

DEVINL void mma8(v8f acc[8], const v16bf a[2], const v16bf b[4]) {
#pragma unroll
  for (int mt = 0; mt < 2; ++mt)
#pragma unroll
    for (int j = 0; j < 4; ++j)
      acc[mt * 4 + j] = wmma_bf16(a[mt], b[j], acc[mt * 4 + j]);
}

// 32 tokens x 64 features per wave, K = E_ = 512.
// Ping-pong buffers, fully unrolled: no register rotation copies.
DEVINL void gemm_32x64(const __bf16* __restrict__ Abase,
                       const __bf16* __restrict__ Bbase, int lane, v8f acc[8]) {
  v16bf a0[2], b0[4], a1[2], b1[4];
  load_a2(a0, Abase, 0, lane);
  load_b4(b0, Bbase, 0, lane);
#pragma unroll
  for (int kk = 0; kk < E_; kk += 64) {
    load_a2(a1, Abase, kk + 32, lane);
    load_b4(b1, Bbase, kk + 32, lane);
    mma8(acc, a0, b0);
    if (kk + 64 < E_) {
      load_a2(a0, Abase, kk + 64, lane);
      load_b4(b0, Bbase, kk + 64, lane);
    }
    mma8(acc, a1, b1);
  }
}

// ---------------- fp32 -> bf16 conversion ----------------
__global__ __launch_bounds__(256) void cvt_bf16(const float* __restrict__ src,
                                                __bf16* __restrict__ dst, int nelem) {
  int i = blockIdx.x * blockDim.x + threadIdx.x;
  int stride = gridDim.x * blockDim.x;
  for (; i < nelem; i += stride) dst[i] = (__bf16)src[i];
}

// ---------------- fused QKV projection + bias + scale + RoPE ----------------
// grid: (N/256, H, 3)  block: 256 (8 waves; wave -> 32 tokens x 64 feats)
__global__ __launch_bounds__(256) void qkv_rope_kernel(
    const __bf16* __restrict__ X,
    const __bf16* __restrict__ Wqb, const __bf16* __restrict__ Wkb,
    const __bf16* __restrict__ Wvb,
    const float* __restrict__ bq, const float* __restrict__ bk,
    const float* __restrict__ bv,
    __bf16* __restrict__ Q, __bf16* __restrict__ Kd, __bf16* __restrict__ Vt) {
  const int lane = threadIdx.x & 31;
  const int wave = threadIdx.x >> 5;
  const int n = lane & 15, hf = lane >> 4;
  const int z = blockIdx.z;                    // 0=q 1=k 2=v
  const int mbase = blockIdx.x * 256 + wave * 32;
  const int h = blockIdx.y;
  const int obase = h * HD_;

  const __bf16* W   = (z == 0) ? Wqb : (z == 1) ? Wkb : Wvb;
  const float* bias = (z == 0) ? bq  : (z == 1) ? bk  : bv;

  v8f acc[8];
#pragma unroll
  for (int i = 0; i < 8; ++i) acc[i] = vzero8();

  gemm_32x64(X + mbase * E_, W + obase * E_, lane, acc);

  const float scale = (z == 0) ? 0.125f : 1.0f;  // HD^-0.5 applied pre-RoPE (linear)
#pragma unroll
  for (int j = 0; j < 4; ++j) {
    const float bj = bias[obase + j * 16 + n];
#pragma unroll
    for (int mt = 0; mt < 2; ++mt)
#pragma unroll
      for (int r = 0; r < 8; ++r)
        acc[mt * 4 + j][r] = (acc[mt * 4 + j][r] + bj) * scale;
  }

  if (z < 2) {
    // RoPE: pair (d, d+32) = (acc[mt*4+j], acc[mt*4+j+2]) same lane/slot.
#pragma unroll
    for (int j = 0; j < 2; ++j) {
      const int d = j * 16 + n;  // 0..31
      const float inv = __expf(-(float)d * 0.28782313662425574f);  // 10000^(-d/32)
#pragma unroll
      for (int mt = 0; mt < 2; ++mt)
#pragma unroll
        for (int r = 0; r < 8; ++r) {
          const int tok = mbase + mt * 16 + r + 8 * hf;
          const float t = (float)(tok >> 2);  // tok = t*B + b, B=4
          float sn, cs;
          __sincosf(t * inv, &sn, &cs);
          const float x1 = acc[mt * 4 + j][r], x2 = acc[mt * 4 + j + 2][r];
          acc[mt * 4 + j][r]     = x1 * cs - x2 * sn;
          acc[mt * 4 + j + 2][r] = x2 * cs + x1 * sn;
        }
    }
  }

  __bf16* out = (z == 0) ? Q : (z == 1) ? Kd : Vt;
#pragma unroll
  for (int mt = 0; mt < 2; ++mt)
#pragma unroll
    for (int j = 0; j < 4; ++j)
#pragma unroll
      for (int r = 0; r < 8; ++r) {
        const int tok = mbase + mt * 16 + r + 8 * hf;
        const int t = tok >> 2, b = tok & 3;
        const int bh = b * H_ + h;
        const int d = j * 16 + n;
        const float v = acc[mt * 4 + j][r];
        if (z < 2) out[(bh * T_ + t) * HD_ + d] = (__bf16)v;   // Q/K: [bh][t][d]
        else       out[(bh * HD_ + d) * T_ + t] = (__bf16)v;   // V^T: [bh][d][t]
      }
}

// ---------------- flash attention ----------------
// grid: (T/64, BH)  block: 128 (4 waves; wave -> 16 q rows x HD)
__global__ __launch_bounds__(128) void flash_attn_kernel(
    const __bf16* __restrict__ Q, const __bf16* __restrict__ Kd,
    const __bf16* __restrict__ Vt, __bf16* __restrict__ ctx) {
  __shared__ __align__(16) __bf16 Plds[4][16 * 64];  // per-wave P tile (bf16)

  const int lane = threadIdx.x & 31;
  const int wave = threadIdx.x >> 5;
  const int n = lane & 15, hf = lane >> 4;
  const int bh = blockIdx.y;
  const int qbase = blockIdx.x * 64 + wave * 16;

  const __bf16* Qb = Q  + (size_t)bh * T_ * HD_;
  const __bf16* Kb = Kd + (size_t)bh * T_ * HD_;
  const __bf16* Vb = Vt + (size_t)bh * HD_ * T_;

  v16bf aq[2];
#pragma unroll
  for (int kk = 0; kk < 2; ++kk)
    aq[kk] = load_frag_a(Qb + qbase * HD_ + kk * 32, HD_, lane);

  v8f oacc[4];
#pragma unroll
  for (int j = 0; j < 4; ++j) oacc[j] = vzero8();
  float mst[8], lst[8];
#pragma unroll
  for (int r = 0; r < 8; ++r) { mst[r] = -1.0e30f; lst[r] = 0.0f; }

  for (int kt = 0; kt < T_; kt += 64) {
    // Prefetch next kv tile: K = contiguous 8KB (64 rows x 128B), V = 64 rows,
    // each row's 64-col slice = one 128B line.  -> global_prefetch_b8
    if (kt + 64 < T_) {
      const __bf16* kp = Kb + (kt + 64) * HD_;
      __builtin_prefetch(kp + lane * 128, 0, 3);          // 32 lanes x 256B
      __builtin_prefetch(kp + lane * 128 + 64, 0, 3);
      __builtin_prefetch(Vb + (size_t)lane * T_ + kt + 64, 0, 3);
      __builtin_prefetch(Vb + (size_t)(lane + 32) * T_ + kt + 64, 0, 3);
    }

    // Batch-load all 8 K fragments (one clause), then 8 back-to-back S WMMAs.
    v16bf kf[8];
#pragma unroll
    for (int j = 0; j < 4; ++j)
#pragma unroll
      for (int kk = 0; kk < 2; ++kk)
        kf[j * 2 + kk] = load_frag_b(Kb + (kt + j * 16) * HD_ + kk * 32, HD_, lane);

    v8f s[4];
#pragma unroll
    for (int j = 0; j < 4; ++j) s[j] = vzero8();
#pragma unroll
    for (int j = 0; j < 4; ++j)
#pragma unroll
      for (int kk = 0; kk < 2; ++kk)
        s[j] = wmma_bf16(aq[kk], kf[j * 2 + kk], s[j]);

    // online softmax (rows live in 16-lane halves -> xor masks 1,2,4,8 only)
    float corr[8];
#pragma unroll
    for (int r = 0; r < 8; ++r) {
      float mx = fmaxf(fmaxf(s[0][r], s[1][r]), fmaxf(s[2][r], s[3][r]));
#pragma unroll
      for (int msk = 1; msk <= 8; msk <<= 1) mx = fmaxf(mx, __shfl_xor(mx, msk, 32));
      const float mnew = fmaxf(mst[r], mx);
      corr[r] = __expf(mst[r] - mnew);
      mst[r] = mnew;
    }
#pragma unroll
    for (int j = 0; j < 4; ++j)
#pragma unroll
      for (int r = 0; r < 8; ++r) s[j][r] = __expf(s[j][r] - mst[r]);
#pragma unroll
    for (int r = 0; r < 8; ++r) {
      float sm = s[0][r] + s[1][r] + s[2][r] + s[3][r];
#pragma unroll
      for (int msk = 1; msk <= 8; msk <<= 1) sm += __shfl_xor(sm, msk, 32);
      lst[r] = lst[r] * corr[r] + sm;
    }
#pragma unroll
    for (int j = 0; j < 4; ++j)
#pragma unroll
      for (int r = 0; r < 8; ++r) oacc[j][r] *= corr[r];

    // acc-layout -> A-layout transpose of P via per-wave LDS tile
#pragma unroll
    for (int j = 0; j < 4; ++j)
#pragma unroll
      for (int r = 0; r < 8; ++r) {
        const int m = r + 8 * hf;
        Plds[wave][m * 64 + j * 16 + n] = (__bf16)s[j][r];
      }

    // Issue all 8 V fragments now: their latency hides under the LDS round-trip.
    v16bf vf[8];
#pragma unroll
    for (int kk = 0; kk < 2; ++kk)
#pragma unroll
      for (int j = 0; j < 4; ++j)
        vf[kk * 4 + j] = load_frag_b(Vb + (j * 16) * T_ + kt + kk * 32, T_, lane);

    asm volatile("s_wait_dscnt 0" ::: "memory");  // wave-local LDS store->load order

    // O += P @ V
#pragma unroll
    for (int kk = 0; kk < 2; ++kk) {
      v16bf ap = load_frag_a(&Plds[wave][0] + kk * 32, 64, lane);
#pragma unroll
      for (int j = 0; j < 4; ++j)
        oacc[j] = wmma_bf16(ap, vf[kk * 4 + j], oacc[j]);
    }
    asm volatile("s_wait_dscnt 0" ::: "memory");  // LDS reads done before overwrite
  }

  // normalize + write context token-major bf16 [N][E]
  const int b = bh >> 3, h = bh & 7;
#pragma unroll
  for (int j = 0; j < 4; ++j)
#pragma unroll
    for (int r = 0; r < 8; ++r) {
      const int m = r + 8 * hf;
      const int t = qbase + m;
      const float ov = oacc[j][r] / lst[r];
      ctx[(size_t)(t * B_ + b) * E_ + h * HD_ + j * 16 + n] = (__bf16)ov;
    }
}

// ---------------- output projection ----------------
// grid: (N/256, E/64)  block: 256
__global__ __launch_bounds__(256) void out_proj_kernel(
    const __bf16* __restrict__ CTX, const __bf16* __restrict__ Wob,
    const float* __restrict__ bo, float* __restrict__ out) {
  const int lane = threadIdx.x & 31;
  const int wave = threadIdx.x >> 5;
  const int n = lane & 15, hf = lane >> 4;
  const int mbase = blockIdx.x * 256 + wave * 32;
  const int obase = blockIdx.y * 64;

  v8f acc[8];
#pragma unroll
  for (int i = 0; i < 8; ++i) acc[i] = vzero8();

  gemm_32x64(CTX + mbase * E_, Wob + obase * E_, lane, acc);

#pragma unroll
  for (int j = 0; j < 4; ++j) {
    const float bj = bo[obase + j * 16 + n];
#pragma unroll
    for (int mt = 0; mt < 2; ++mt)
#pragma unroll
      for (int r = 0; r < 8; ++r) {
        const int tok = mbase + mt * 16 + r + 8 * hf;
        out[(size_t)tok * E_ + obase + j * 16 + n] = acc[mt * 4 + j][r] + bj;
      }
  }
}

extern "C" void kernel_launch(void* const* d_in, const int* in_sizes, int n_in,
                              void* d_out, int out_size, void* d_ws, size_t ws_size,
                              hipStream_t stream) {
  (void)in_sizes; (void)n_in; (void)out_size; (void)ws_size;
  const float* query = (const float*)d_in[0];
  const float* Wq = (const float*)d_in[1];
  const float* bq = (const float*)d_in[2];
  const float* Wk = (const float*)d_in[3];
  const float* bk = (const float*)d_in[4];
  const float* Wv = (const float*)d_in[5];
  const float* bv = (const float*)d_in[6];
  const float* Wo = (const float*)d_in[7];
  const float* bo = (const float*)d_in[8];
  float* out = (float*)d_out;

  char* w = (char*)d_ws;
  size_t off = 0;
  auto alloc = [&](size_t bytes) {
    char* p = w + off;
    off += (bytes + 255) & ~(size_t)255;
    return p;
  };
  __bf16* Xbf = (__bf16*)alloc((size_t)N_ * E_ * 2);   // 8 MB
  __bf16* Wqb = (__bf16*)alloc((size_t)E_ * E_ * 2);   // 512 KB each
  __bf16* Wkb = (__bf16*)alloc((size_t)E_ * E_ * 2);
  __bf16* Wvb = (__bf16*)alloc((size_t)E_ * E_ * 2);
  __bf16* Wob = (__bf16*)alloc((size_t)E_ * E_ * 2);
  __bf16* Qw  = (__bf16*)alloc((size_t)BH_ * T_ * HD_ * 2);  // 8 MB
  __bf16* Kw  = (__bf16*)alloc((size_t)BH_ * T_ * HD_ * 2);  // 8 MB
  __bf16* Vtw = (__bf16*)alloc((size_t)BH_ * HD_ * T_ * 2);  // 8 MB (transposed)
  __bf16* CTX = (__bf16*)alloc((size_t)N_ * E_ * 2);         // 8 MB

  cvt_bf16<<<(N_ * E_ + 255) / 256, 256, 0, stream>>>(query, Xbf, N_ * E_);
  cvt_bf16<<<(E_ * E_ + 255) / 256, 256, 0, stream>>>(Wq, Wqb, E_ * E_);
  cvt_bf16<<<(E_ * E_ + 255) / 256, 256, 0, stream>>>(Wk, Wkb, E_ * E_);
  cvt_bf16<<<(E_ * E_ + 255) / 256, 256, 0, stream>>>(Wv, Wvb, E_ * E_);
  cvt_bf16<<<(E_ * E_ + 255) / 256, 256, 0, stream>>>(Wo, Wob, E_ * E_);

  qkv_rope_kernel<<<dim3(N_ / 256, H_, 3), 256, 0, stream>>>(
      Xbf, Wqb, Wkb, Wvb, bq, bk, bv, Qw, Kw, Vtw);

  flash_attn_kernel<<<dim3(T_ / 64, BH_), 128, 0, stream>>>(Qw, Kw, Vtw, CTX);

  out_proj_kernel<<<dim3(N_ / 256, E_ / 64), 256, 0, stream>>>(CTX, Wob, bo, out);
}
